// CSOCSSC_v41_11287174054533
// MI455X (gfx1250) — compile-verified, matrix operand order checked by reading, not run on hardware
//
#include <hip/hip_runtime.h>
#include <hip/hip_bf16.h>

#define N_NODES 50000
#define N_EDGES 800000
#define ND 64
#define HD 128
#define ED 32
#define IN_DIM 160            // 2*ND + ED
#define EPB 128               // edges per block (8 waves x 16-edge M-tiles)
#define MSTRIDE 168           // bf16 elems per LDS row (336B, 16B aligned)
#define H_OUT_ELEMS (N_NODES * ND)

typedef __bf16 v16bf __attribute__((ext_vector_type(16)));
typedef float  v8f   __attribute__((ext_vector_type(8)));
typedef unsigned int v4u __attribute__((ext_vector_type(4)));

union FragBF { v16bf bf; v4u u[2]; };

__device__ __forceinline__ float silu_f(float v) {
    return v / (1.0f + __expf(-v));
}

// ---- packed bf16 weight fragments in d_ws (B-matrix lane layout) ----
// fragment = 512 bf16, laid out lane-major: elem(lane, i) at frag*512 + lane*16 + i
// lane -> n = (lane&15), hi = lane>>4 ; i -> K = i + 8*hi + (i>=8 ? 8 : 0)
#define FRAGS_WN1 40   /* 5 ksteps x 8 ntiles (160x128) */
#define FRAGS_WC1 40
#define FRAGS_WN2 16   /* 4 ksteps x 4 ntiles (128x64) */
#define PACK_WN1_OFF 0
#define PACK_WC1_OFF (FRAGS_WN1 * 512)
#define PACK_WN2_OFF ((FRAGS_WN1 + FRAGS_WC1) * 512)
#define PACK_TOTAL  ((FRAGS_WN1 + FRAGS_WC1 + FRAGS_WN2) * 512)

__global__ void pack_weights_kernel(const float* __restrict__ Wn1,
                                    const float* __restrict__ Wc1,
                                    const float* __restrict__ Wn2,
                                    __bf16* __restrict__ pack) {
    int idx = blockIdx.x * blockDim.x + threadIdx.x;
    if (idx >= PACK_TOTAL) return;
    int frag   = idx >> 9;
    int within = idx & 511;
    int lane = within >> 4;
    int i    = within & 15;
    int hi   = lane >> 4;
    int n16  = lane & 15;
    int K = i + 8 * hi + ((i >= 8) ? 8 : 0);
    float v;
    if (frag < FRAGS_WN1) {
        int ks = frag >> 3, nt = frag & 7;
        v = Wn1[(ks * 32 + K) * HD + nt * 16 + n16];
    } else if (frag < FRAGS_WN1 + FRAGS_WC1) {
        int f = frag - FRAGS_WN1;
        int ks = f >> 3, nt = f & 7;
        v = Wc1[(ks * 32 + K) * HD + nt * 16 + n16];
    } else {
        int f = frag - FRAGS_WN1 - FRAGS_WC1;
        int ks = f >> 2, nt = f & 3;
        v = Wn2[(ks * 32 + K) * ND + nt * 16 + n16];
    }
    pack[idx] = (__bf16)v;
}

__global__ void init_out_kernel(const float* __restrict__ h,
                                const float* __restrict__ x,
                                float* __restrict__ out) {
    int i = blockIdx.x * blockDim.x + threadIdx.x;
    if (i < H_OUT_ELEMS) out[i] = h[i];
    else if (i < H_OUT_ELEMS + N_NODES * 3) out[i] = x[i - H_OUT_ELEMS];
}

__launch_bounds__(256)
__global__ void egnn_edge_kernel(const float* __restrict__ h,
                                 const float* __restrict__ x,
                                 const int*   __restrict__ edge_index,
                                 const float* __restrict__ edge_dist,
                                 const float* __restrict__ We1, const float* __restrict__ be1,
                                 const float* __restrict__ We2, const float* __restrict__ be2,
                                 const float* __restrict__ bn1, const float* __restrict__ bn2,
                                 const float* __restrict__ bc1, const float* __restrict__ Wc2,
                                 const __bf16* __restrict__ pack,
                                 float* __restrict__ out) {
    // m_input tile [128 edges][160] bf16; cols 0..127 of each wave's 16 rows are
    // later reused to hold silu(t_n) [16][128] bf16 (wave-private reuse).
    __shared__ __align__(16) __bf16 sM[EPB * MSTRIDE];
    __shared__ float sWe2[ED * ED];
    __shared__ float sWe1[ED], sbe1[ED], sbe2[ED];
    __shared__ float sbn1[HD], sbc1[HD], sWc2[HD], sbn2[ND];
    __shared__ int   sSrc[EPB], sDst[EPB];
    __shared__ float sCW[EPB];

    const int tid   = threadIdx.x;
    const int ebase = blockIdx.x * EPB;

    // ---- stage edge indices + small constants ----
    if (tid < EPB) {
        sSrc[tid] = edge_index[ebase + tid];             // row 0 = src
        sDst[tid] = edge_index[N_EDGES + ebase + tid];   // row 1 = dst
    }
    if (tid < ED) { sWe1[tid] = We1[tid]; sbe1[tid] = be1[tid]; sbe2[tid] = be2[tid]; }
    for (int i = tid; i < ED * ED; i += 256) sWe2[i] = We2[i];
    if (tid < HD) { sbn1[tid] = bn1[tid]; sbc1[tid] = bc1[tid]; sWc2[tid] = Wc2[tid]; }
    if (tid < ND) sbn2[tid] = bn2[tid];
    __syncthreads();

    // ---- gather h[src], h[dst] into sM cols [0..127] (f32 -> bf16) ----
    // 128 edges x 32 float4 chunks (16 src + 16 dst); h rows are L2-resident.
    for (int c = tid; c < EPB * 32; c += 256) {
        int e = c >> 5, q = c & 31;
        int node = (q < 16) ? sSrc[e] : sDst[e];
        int col  = (q & 15) * 4;
        float4 v = *(const float4*)(h + node * ND + col);
        __bf16* p = &sM[e * MSTRIDE + ((q < 16) ? 0 : ND) + col];
        p[0] = (__bf16)v.x; p[1] = (__bf16)v.y; p[2] = (__bf16)v.z; p[3] = (__bf16)v.w;
    }
    // ---- edge MLP -> sM cols [128..159] ----
    if (tid < EPB) {
        float d = edge_dist[ebase + tid];
        float vb[ED];
        #pragma unroll
        for (int j = 0; j < ED; ++j) vb[j] = silu_f(d * sWe1[j] + sbe1[j]);
        for (int c0 = 0; c0 < ED; ++c0) {
            float a = sbe2[c0];
            #pragma unroll
            for (int j = 0; j < ED; ++j) a += vb[j] * sWe2[j * ED + c0];
            sM[tid * MSTRIDE + 2 * ND + c0] = (__bf16)a;
        }
    }
    __syncthreads();

    const int wave = tid >> 5;
    const int lane = tid & 31;
    const int hi   = lane >> 4;
    const int n16  = lane & 15;
    __bf16* myM = &sM[wave * 16 * MSTRIDE];   // this wave's 16-edge M-tile

    // ======== Pass 1: coord branch  t_c = silu(m_input @ Wc1 + bc1); cw = t_c @ Wc2 ========
    {
        v8f acc[8] = {};
        #pragma unroll
        for (int ks = 0; ks < 5; ++ks) {
            FragBF a;
            const __bf16* ap = myM + n16 * MSTRIDE + ks * 32 + 8 * hi;
            a.u[0] = *(const v4u*)ap;          // K = kbase + 8hi + [0..7]
            a.u[1] = *(const v4u*)(ap + 16);   // K = kbase + 16 + 8hi + [0..7]
            #pragma unroll
            for (int nt = 0; nt < 8; ++nt) {
                FragBF b;
                const __bf16* bp = pack + PACK_WC1_OFF + (ks * 8 + nt) * 512 + lane * 16;
                b.u[0] = *(const v4u*)bp;
                b.u[1] = *(const v4u*)(bp + 8);
                acc[nt] = __builtin_amdgcn_wmma_f32_16x16x32_bf16(
                    false, a.bf, false, b.bf, (short)0, acc[nt], false, false);
            }
        }
        // silu + dot with Wc2 in C-layout; reduce over the 16 lanes holding one row
        #pragma unroll
        for (int r = 0; r < 8; ++r) {
            float s = 0.0f;
            #pragma unroll
            for (int nt = 0; nt < 8; ++nt) {
                int n = n16 + 16 * nt;
                s += silu_f(acc[nt][r] + sbc1[n]) * sWc2[n];
            }
            s += __shfl_xor(s, 1, 32);
            s += __shfl_xor(s, 2, 32);
            s += __shfl_xor(s, 4, 32);
            s += __shfl_xor(s, 8, 32);
            if (n16 == 0) sCW[wave * 16 + r + 8 * hi] = s;   // row m = r + 8*hi
        }
    }

    // ======== Pass 2: node branch  t_n = silu(m_input @ Wn1 + bn1) -> myM cols [0..127] ========
    {
        v8f acc[8] = {};
        #pragma unroll
        for (int ks = 0; ks < 5; ++ks) {
            FragBF a;
            const __bf16* ap = myM + n16 * MSTRIDE + ks * 32 + 8 * hi;
            a.u[0] = *(const v4u*)ap;
            a.u[1] = *(const v4u*)(ap + 16);
            #pragma unroll
            for (int nt = 0; nt < 8; ++nt) {
                FragBF b;
                const __bf16* bp = pack + PACK_WN1_OFF + (ks * 8 + nt) * 512 + lane * 16;
                b.u[0] = *(const v4u*)bp;
                b.u[1] = *(const v4u*)(bp + 8);
                acc[nt] = __builtin_amdgcn_wmma_f32_16x16x32_bf16(
                    false, a.bf, false, b.bf, (short)0, acc[nt], false, false);
            }
        }
        // silu -> bf16, overwrite this wave's own LDS rows (all reads above are done)
        #pragma unroll
        for (int nt = 0; nt < 8; ++nt) {
            #pragma unroll
            for (int r = 0; r < 8; ++r) {
                int n = n16 + 16 * nt;
                int m = r + 8 * hi;
                myM[m * MSTRIDE + n] = (__bf16)silu_f(acc[nt][r] + sbn1[n]);
            }
        }
    }

    // ======== GEMM2: m = t_n @ Wn2 + bn2 ; scatter-add into out by dst ========
    {
        v8f acc[4] = {};
        #pragma unroll
        for (int ks = 0; ks < 4; ++ks) {
            FragBF a;
            const __bf16* ap = myM + n16 * MSTRIDE + ks * 32 + 8 * hi;
            a.u[0] = *(const v4u*)ap;
            a.u[1] = *(const v4u*)(ap + 16);
            #pragma unroll
            for (int nt = 0; nt < 4; ++nt) {
                FragBF b;
                const __bf16* bp = pack + PACK_WN2_OFF + (ks * 4 + nt) * 512 + lane * 16;
                b.u[0] = *(const v4u*)bp;
                b.u[1] = *(const v4u*)(bp + 8);
                acc[nt] = __builtin_amdgcn_wmma_f32_16x16x32_bf16(
                    false, a.bf, false, b.bf, (short)0, acc[nt], false, false);
            }
        }
        #pragma unroll
        for (int nt = 0; nt < 4; ++nt) {
            #pragma unroll
            for (int r = 0; r < 8; ++r) {
                int n = n16 + 16 * nt;
                int m = r + 8 * hi;
                float val = acc[nt][r] + sbn2[n];
                atomicAdd(&out[sDst[wave * 16 + m] * ND + n], val);
            }
        }
    }

    // ======== coord update scatter (3 floats/edge) ========
    if (lane < 16) {
        int m = wave * 16 + lane;
        int s = sSrc[m], d = sDst[m];
        float dx = x[s * 3 + 0] - x[d * 3 + 0];
        float dy = x[s * 3 + 1] - x[d * 3 + 1];
        float dz = x[s * 3 + 2] - x[d * 3 + 2];
        float len = sqrtf(dx * dx + dy * dy + dz * dz);
        len = fmaxf(len, 1e-8f);
        float w = sCW[m] / len;
        float* xo = out + H_OUT_ELEMS;
        atomicAdd(&xo[d * 3 + 0], w * dx);
        atomicAdd(&xo[d * 3 + 1], w * dy);
        atomicAdd(&xo[d * 3 + 2], w * dz);
    }
}

extern "C" void kernel_launch(void* const* d_in, const int* in_sizes, int n_in,
                              void* d_out, int out_size, void* d_ws, size_t ws_size,
                              hipStream_t stream) {
    (void)in_sizes; (void)n_in; (void)out_size; (void)ws_size;
    const float* h          = (const float*)d_in[0];
    const float* x          = (const float*)d_in[1];
    const int*   edge_index = (const int*)d_in[2];
    const float* edge_dist  = (const float*)d_in[3];
    const float* We1 = (const float*)d_in[4];
    const float* be1 = (const float*)d_in[5];
    const float* We2 = (const float*)d_in[6];
    const float* be2 = (const float*)d_in[7];
    const float* Wn1 = (const float*)d_in[8];
    const float* bn1 = (const float*)d_in[9];
    const float* Wn2 = (const float*)d_in[10];
    const float* bn2 = (const float*)d_in[11];
    const float* Wc1 = (const float*)d_in[12];
    const float* bc1 = (const float*)d_in[13];
    const float* Wc2 = (const float*)d_in[14];
    float*  out  = (float*)d_out;
    __bf16* pack = (__bf16*)d_ws;

    pack_weights_kernel<<<(PACK_TOTAL + 255) / 256, 256, 0, stream>>>(Wn1, Wc1, Wn2, pack);

    int tot = H_OUT_ELEMS + N_NODES * 3;
    init_out_kernel<<<(tot + 255) / 256, 256, 0, stream>>>(h, x, out);

    egnn_edge_kernel<<<N_EDGES / EPB, 256, 0, stream>>>(
        h, x, edge_index, edge_dist,
        We1, be1, We2, be2, bn1, bn2, bc1, Wc2, pack, out);
}
